// transformer_41077067219265
// MI455X (gfx1250) — matrix operand order repaired and checked by reading, not compiled, not run on hardware
//
#include <hip/hip_runtime.h>
#include <math.h>

// ---------------------------------------------------------------------------
// MI455X (gfx1250) channel-attention block.
// B=16, C=384, N=H*W=4096, HEADS=8, ch=48.
// Big GEMMs run on v_wmma_f32_16x16x32_bf16 (wave32 WMMA). L2 normalization is
// folded into a post-GEMM scale of the raw Gram matrix (mathematically equal).
// x is pre-transposed/converted to bf16 [b][n][c] so GEMM staging is a pure
// 128-bit copy into LDS in the WMMA B-fragment-friendly layout.
// ---------------------------------------------------------------------------

typedef __bf16 bf16_t;
typedef bf16_t v16bf __attribute__((ext_vector_type(16)));
typedef float  v8f   __attribute__((ext_vector_type(8)));

union Frag {
    uint4 q[2];   // 32 bytes = 16 bf16
    v16bf v;
};

__device__ __forceinline__ unsigned short f2bf(float f) {
    unsigned int u = __float_as_uint(f);
    u += 0x7FFFu + ((u >> 16) & 1u);      // round-to-nearest-even
    return (unsigned short)(u >> 16);
}
__device__ __forceinline__ float bf2f(unsigned short u) {
    return __uint_as_float(((unsigned int)u) << 16);
}

// ---------------------------------------------------------------------------
// Kernel 0a: fp32 -> bf16 weight conversion (w is [384][384] row-major o,c).
// ---------------------------------------------------------------------------
__global__ void cvt_bf16_kernel(const float* __restrict__ w,
                                unsigned short* __restrict__ o, int n) {
    int i = blockIdx.x * 256 + threadIdx.x;
    if (i < n) o[i] = f2bf(w[i]);
}

// ---------------------------------------------------------------------------
// Kernel 0b: x[b][c][n] fp32 -> xT[b][n][c] bf16 (LDS-tiled 64x64 transpose,
// coalesced on both sides).  Grid: (N/64, C/64, B).  Block: 256.
// ---------------------------------------------------------------------------
__global__ __launch_bounds__(256) void xpose_kernel(
    const float* __restrict__ x, unsigned short* __restrict__ xt)
{
    const int n0 = blockIdx.x * 64;
    const int c0 = blockIdx.y * 64;
    const int b  = blockIdx.z;
    const int tid = threadIdx.x;
    __shared__ unsigned short sT[64 * 65];

#pragma unroll
    for (int i = 0; i < 16; ++i) {
        int idx = i * 256 + tid;                 // 4096 = 64c x 64n
        int ci = idx >> 6, nj = idx & 63;
        float v = x[(((size_t)(b * 384 + c0 + ci)) << 12) + n0 + nj];
        sT[nj * 65 + ci] = f2bf(v);
    }
    __syncthreads();
#pragma unroll
    for (int i = 0; i < 16; ++i) {
        int idx = i * 256 + tid;
        int nj = idx >> 6, ci = idx & 63;
        xt[((size_t)(b * 4096 + n0 + nj)) * 384 + c0 + ci] = sT[nj * 65 + ci];
    }
}

// ---------------------------------------------------------------------------
// Kernel 1: QKV projection GEMM.  q/k/v[o][n] = sum_c W[o][c] * xT[n][c] + b.
// Grid: (C/64, N/128, B).  Block: 256 threads = 8 waves (4 row-waves x 2 col).
// ---------------------------------------------------------------------------
__global__ __launch_bounds__(256) void qkv_kernel(
    const unsigned short* __restrict__ xt,
    const unsigned short* __restrict__ wq,
    const unsigned short* __restrict__ wk,
    const unsigned short* __restrict__ wv,
    const float* __restrict__ b1, const float* __restrict__ b2,
    const float* __restrict__ b3,
    unsigned short* __restrict__ qb, unsigned short* __restrict__ kb,
    float* __restrict__ vf)
{
    const int ctile = blockIdx.x;            // 0..5   (64 output rows)
    const int ntile = blockIdx.y;            // 0..31  (128 cols)
    const int b     = blockIdx.z;            // 0..15
    const int tid   = threadIdx.x;
    const int lane  = tid & 31;
    const int wave  = tid >> 5;
    const int wm    = wave & 3;              // row wave (16 rows each)
    const int wn    = wave >> 2;             // col wave (64 cols each)
    const int rowbase = ctile * 64 + wm * 16;
    const int n0      = ntile * 128;
    const int colbase = wn * 64;

    // x chunk staged: sh[n][k], padded stride 40 ushorts (bank spread)
    __shared__ unsigned short sh[128 * 40];

    v8f acc[3][4];
#pragma unroll
    for (int i = 0; i < 3; ++i)
#pragma unroll
        for (int s = 0; s < 4; ++s) acc[i][s] = (v8f){};

    // 16-bit A-matrix lane layout: lanes 0-15 take K offsets {0..7,16..23},
    // lanes 16-31 take {8..15,24..31}, row = lane&15.
    const int c0    = (lane & 16) ? 8 : 0;
    const int rA    = rowbase + (lane & 15);
    const int khalf = lane >> 4;

    // staging addressing: 512 16-byte granules (128 n-rows x 4 parts)
    const int sgN    = (2 * 256 > 512) ? 0 : 0; (void)sgN;

    for (int kk = 0; kk < 384; kk += 32) {
        __syncthreads();
        // ---- stage xT[b][n0..n0+128)[kk..kk+32) -> LDS (pure b128 copy) ----
#pragma unroll
        for (int i = 0; i < 2; ++i) {
            int idx  = i * 256 + tid;            // 0..511
            int n    = idx >> 2;
            int part = (idx & 3) * 8;            // 8 bf16 per granule
            const uint4 src = *(const uint4*)(
                xt + ((size_t)(b * 4096 + n0 + n)) * 384 + kk + part);
            *(uint4*)(sh + n * 40 + part) = src;
        }
        __syncthreads();

        if (kk + 32 < 384) {   // hint next weight chunk -> global_prefetch
            __builtin_prefetch(wq + (size_t)rA * 384 + kk + 32, 0, 3);
        }

        // ---- A fragments (weights, contiguous 16B chunks) ----
        Frag aq, ak, av;
        size_t aoff = (size_t)rA * 384 + kk + c0;
        aq.q[0] = *(const uint4*)(wq + aoff);
        aq.q[1] = *(const uint4*)(wq + aoff + 16);
        ak.q[0] = *(const uint4*)(wk + aoff);
        ak.q[1] = *(const uint4*)(wk + aoff + 16);
        av.q[0] = *(const uint4*)(wv + aoff);
        av.q[1] = *(const uint4*)(wv + aoff + 16);

        // ---- B fragments from LDS: lane = col n, K {c0..c0+7, c0+16..c0+23} ----
        Frag bfr[4];
#pragma unroll
        for (int s = 0; s < 4; ++s) {
            int nc = colbase + s * 16 + (lane & 15);
            int sb = nc * 40 + c0;
            bfr[s].q[0] = *(const uint4*)(sh + sb);
            bfr[s].q[1] = *(const uint4*)(sh + sb + 16);
        }

#pragma unroll
        for (int s = 0; s < 4; ++s) {
            acc[0][s] = __builtin_amdgcn_wmma_f32_16x16x32_bf16(
                false, aq.v, false, bfr[s].v, (short)0, acc[0][s], false, false);
            acc[1][s] = __builtin_amdgcn_wmma_f32_16x16x32_bf16(
                false, ak.v, false, bfr[s].v, (short)0, acc[1][s], false, false);
            acc[2][s] = __builtin_amdgcn_wmma_f32_16x16x32_bf16(
                false, av.v, false, bfr[s].v, (short)0, acc[2][s], false, false);
        }
    }

    // ---- epilogue: bias, store q/k bf16 and v fp32 ----
    const int rofs = (lane & 16) ? 8 : 0;    // C/D layout: VGPR r -> row r / r+8
#pragma unroll
    for (int s = 0; s < 4; ++s) {
        int col = n0 + colbase + s * 16 + (lane & 15);
#pragma unroll
        for (int r = 0; r < 8; ++r) {
            int row = rowbase + r + rofs;
            size_t o = (((size_t)(b * 384 + row)) << 12) + col;
            qb[o] = f2bf(acc[0][s][r] + b1[row]);
            kb[o] = f2bf(acc[1][s][r] + b2[row]);
            vf[o] = acc[2][s][r] + b3[row];
        }
    }
}

// ---------------------------------------------------------------------------
// Kernel 2: per-channel sum of squares (for L2 norms).
// Grid: (B*C, 2) -- y=0: q, y=1: k.  Block: 256.
// ---------------------------------------------------------------------------
__global__ __launch_bounds__(256) void sumsq_kernel(
    const unsigned short* __restrict__ qb,
    const unsigned short* __restrict__ kb,
    float* __restrict__ qss, float* __restrict__ kss)
{
    const int bc  = blockIdx.x;
    const int sel = blockIdx.y;
    const unsigned short* p = sel ? kb : qb;
    const uint4* pv = (const uint4*)(p + ((size_t)bc << 12));
    const int tid = threadIdx.x, lane = tid & 31, wave = tid >> 5;

    float s = 0.f;
#pragma unroll
    for (int i = 0; i < 2; ++i) {
        uint4 u = pv[tid + i * 256];          // 8 bf16
        unsigned int w[4] = {u.x, u.y, u.z, u.w};
#pragma unroll
        for (int j = 0; j < 4; ++j) {
            float a = bf2f((unsigned short)(w[j] & 0xFFFF));
            float c = bf2f((unsigned short)(w[j] >> 16));
            s += a * a + c * c;
        }
    }
#pragma unroll
    for (int off = 16; off > 0; off >>= 1) s += __shfl_down(s, off);
    __shared__ float red[8];
    if (lane == 0) red[wave] = s;
    __syncthreads();
    if (tid == 0) {
        float t = 0.f;
#pragma unroll
        for (int w = 0; w < 8; ++w) t += red[w];
        (sel ? kss : qss)[bc] = t;
    }
}

// ---------------------------------------------------------------------------
// Kernel 3: attention matrix per (b,h): S = q@k^T over N=4096 (bf16 WMMA),
// scale by 1/(|q_c||k_d|)*temp, softmax rows. Block: 128 thr, waves split K.
// ---------------------------------------------------------------------------
__global__ __launch_bounds__(128) void attn_kernel(
    const unsigned short* __restrict__ qb,
    const unsigned short* __restrict__ kb,
    const float* __restrict__ qss, const float* __restrict__ kss,
    const float* __restrict__ temp, float* __restrict__ attn)
{
    const int b = blockIdx.x >> 3, h = blockIdx.x & 7;
    const int tid = threadIdx.x, lane = tid & 31, wave = tid >> 5; // 4 waves
    const size_t base = ((size_t)(b * 384 + h * 48)) << 12;

    v8f acc[3][3];
#pragma unroll
    for (int i = 0; i < 3; ++i)
#pragma unroll
        for (int j = 0; j < 3; ++j) acc[i][j] = (v8f){};

    const int c0    = (lane & 16) ? 8 : 0;
    const int khalf = lane >> 4;
    const int kstart = wave * 1024;

    for (int kk = kstart; kk < kstart + 1024; kk += 32) {
        Frag a[3], bb[3];
#pragma unroll
        for (int i = 0; i < 3; ++i) {
            size_t off = base + (((size_t)(i * 16 + (lane & 15))) << 12) + kk;
            a[i].q[0]  = *(const uint4*)(qb + off + c0);
            a[i].q[1]  = *(const uint4*)(qb + off + c0 + 16);
            bb[i].q[0] = *(const uint4*)(kb + off + c0);
            bb[i].q[1] = *(const uint4*)(kb + off + c0 + 16);
        }
#pragma unroll
        for (int i = 0; i < 3; ++i)
#pragma unroll
            for (int j = 0; j < 3; ++j)
                acc[i][j] = __builtin_amdgcn_wmma_f32_16x16x32_bf16(
                    false, a[i].v, false, bb[j].v, (short)0, acc[i][j],
                    false, false);
    }

    __shared__ float sPart[4][48 * 48];
    __shared__ float sS[48 * 48];
    __shared__ float sRq[48], sRk[48];
    const int rofs = (lane & 16) ? 8 : 0;
#pragma unroll
    for (int i = 0; i < 3; ++i)
#pragma unroll
        for (int j = 0; j < 3; ++j)
#pragma unroll
            for (int r = 0; r < 8; ++r) {
                int row = i * 16 + r + rofs;
                int col = j * 16 + (lane & 15);
                sPart[wave][row * 48 + col] = acc[i][j][r];
            }
    if (tid < 48) {
        sRq[tid] = 1.f / fmaxf(sqrtf(qss[b * 384 + h * 48 + tid]), 1e-12f);
        sRk[tid] = 1.f / fmaxf(sqrtf(kss[b * 384 + h * 48 + tid]), 1e-12f);
    }
    const float tmp = temp[h];
    __syncthreads();

    for (int e = tid; e < 2304; e += 128) {
        float s = sPart[0][e] + sPart[1][e] + sPart[2][e] + sPart[3][e];
        int row = e / 48, col = e - row * 48;
        sS[e] = s * sRq[row] * sRk[col] * tmp;
    }
    __syncthreads();

    if (tid < 48) {
        float m = -1e30f;
        for (int d = 0; d < 48; ++d) m = fmaxf(m, sS[tid * 48 + d]);
        float sum = 0.f;
        for (int d = 0; d < 48; ++d) {
            float e = expf(sS[tid * 48 + d] - m);
            sS[tid * 48 + d] = e;
            sum += e;
        }
        float r = 1.f / sum;
        float* ao = attn + (size_t)((b * 8 + h) * 48 + tid) * 48;
        for (int d = 0; d < 48; ++d) ao[d] = sS[tid * 48 + d] * r;
    }
}

// ---------------------------------------------------------------------------
// Kernel 4: out = attn @ v + v + x  (fp32, memory bound; 48-deep dots).
// Grid: (N/256, H, B).  Block: 256 (one spatial column per thread).
// ---------------------------------------------------------------------------
__global__ __launch_bounds__(256) void out_kernel(
    const float* __restrict__ attn, const float* __restrict__ vf,
    const float* __restrict__ x, float* __restrict__ out)
{
    const int nt = blockIdx.x, h = blockIdx.y, b = blockIdx.z;
    const int tid = threadIdx.x;
    __shared__ float sA[48 * 48];
    const float* ab = attn + (size_t)(b * 8 + h) * 2304;
    for (int e = tid; e < 2304; e += 256) sA[e] = ab[e];
    __syncthreads();

    const int n = nt * 256 + tid;
    const size_t cb = ((size_t)(b * 384 + h * 48)) << 12;

    float vr[48];
#pragma unroll
    for (int d = 0; d < 48; ++d) vr[d] = vf[cb + (((size_t)d) << 12) + n];

    for (int cc = 0; cc < 48; ++cc) {
        const float* ar = &sA[cc * 48];
        float a = 0.f;
#pragma unroll
        for (int d = 0; d < 48; ++d) a = fmaf(ar[d], vr[d], a);
        size_t o = cb + (((size_t)cc) << 12) + n;
        out[o] = a + vf[o] + x[o];       // re-read v[cc] (avoids reg indexing)
    }
}

// ---------------------------------------------------------------------------
extern "C" void kernel_launch(void* const* d_in, const int* in_sizes, int n_in,
                              void* d_out, int out_size, void* d_ws,
                              size_t ws_size, hipStream_t stream) {
    const float* x    = (const float*)d_in[0];
    const float* w1   = (const float*)d_in[1];
    const float* b1   = (const float*)d_in[2];
    const float* w2   = (const float*)d_in[3];
    const float* b2   = (const float*)d_in[4];
    const float* w3   = (const float*)d_in[5];
    const float* b3   = (const float*)d_in[6];
    const float* temp = (const float*)d_in[7];
    float* out        = (float*)d_out;

    unsigned char* p = (unsigned char*)d_ws;
    auto carve = [&](size_t bytes) -> void* {
        void* r = (void*)p;
        p += (bytes + 255) & ~(size_t)255;
        return r;
    };
    const size_t WBYTES  = (size_t)384 * 384 * 2;        // bf16 weights
    const size_t XTBYTES = (size_t)16 * 4096 * 384 * 2;  // bf16 x^T
    const size_t QKBYTES = (size_t)16 * 384 * 4096 * 2;  // bf16 q / k
    const size_t VBYTES  = (size_t)16 * 384 * 4096 * 4;  // fp32 v

    unsigned short* wqb = (unsigned short*)carve(WBYTES);
    unsigned short* wkb = (unsigned short*)carve(WBYTES);
    unsigned short* wvb = (unsigned short*)carve(WBYTES);
    unsigned short* xtb = (unsigned short*)carve(XTBYTES);
    unsigned short* qb  = (unsigned short*)carve(QKBYTES);
    unsigned short* kb  = (unsigned short*)carve(QKBYTES);
    float* vfb  = (float*)carve(VBYTES);
    float* qss  = (float*)carve((size_t)16 * 384 * 4);
    float* kss  = (float*)carve((size_t)16 * 384 * 4);
    float* attn = (float*)carve((size_t)16 * 8 * 48 * 48 * 4);

    const int nW = 384 * 384;
    cvt_bf16_kernel<<<(nW + 255) / 256, 256, 0, stream>>>(w1, wqb, nW);
    cvt_bf16_kernel<<<(nW + 255) / 256, 256, 0, stream>>>(w2, wkb, nW);
    cvt_bf16_kernel<<<(nW + 255) / 256, 256, 0, stream>>>(w3, wvb, nW);

    xpose_kernel<<<dim3(64, 6, 16), 256, 0, stream>>>(x, xtb);

    qkv_kernel<<<dim3(6, 32, 16), 256, 0, stream>>>(
        xtb, wqb, wkb, wvb, b1, b2, b3, qb, kb, vfb);

    sumsq_kernel<<<dim3(16 * 384, 2), 256, 0, stream>>>(qb, kb, qss, kss);

    attn_kernel<<<dim3(128), 128, 0, stream>>>(qb, kb, qss, kss, temp, attn);

    out_kernel<<<dim3(16, 8, 16), 256, 0, stream>>>(attn, vfb, x, out);
}
